// Decoder_83373905150119
// MI455X (gfx1250) — matrix operand order repaired and checked by reading, not compiled, not run on hardware
//
#include <hip/hip_runtime.h>
#include <hip/hip_bf16.h>
#include <cstddef>

typedef __bf16 bf16;
typedef __attribute__((ext_vector_type(16))) __bf16 v16bf;
typedef __attribute__((ext_vector_type(8)))  float  v8f;

// ---------------- constants ----------------
#define BATCH   32
#define T_IN    400
#define ENC     512
#define N_MEL   80
#define NF      2
#define T_OUT   600
#define STEPS   300
#define RNN     1024
#define PRENET  256
#define ATT_DIM 128
#define LOC_F   32
#define LOC_K   31

#define MEL_ELEMS   (BATCH * N_MEL * T_OUT)        // 1,536,000
#define GATE_ELEMS  (BATCH * T_OUT)                // 19,200

__device__ __forceinline__ float sigf(float x) { return 1.0f / (1.0f + __expf(-x)); }

// ---- WMMA fragment loaders (bf16, wave32) ----
// A 16x32 (16-bit): lanes 0-15 row M=lane, elems 0..7 = K k0..k0+7, elems 8..15 = K k0+16..k0+23
//                   lanes 16-31 same rows with k0 shifted by +8 (caller adds grp*8)
__device__ __forceinline__ v16bf load_afrag(const bf16* p) {
  union { uint4 u[2]; v16bf v; } t;
  t.u[0] = *(const uint4*)(p);
  t.u[1] = *(const uint4*)(p + 16);
  return t.v;
}
// B 32x16: lane holds column n = lane&15; same K layout as A (elems 0..7 = k0..k0+7, 8..15 = k0+16..k0+23, caller adds grp*8)
__device__ __forceinline__ v16bf load_bfrag(const bf16* p) {
  union { uint4 u[2]; v16bf v; } t;
  t.u[0] = *(const uint4*)(p);
  t.u[1] = *(const uint4*)(p + 16);
  return t.v;
}

#define WMMA_BF16(A, B, C) \
  __builtin_amdgcn_wmma_f32_16x16x32_bf16(false, (A), false, (B), (short)0, (C), false, false)

// One concat segment of the K loop: flat pointer advance (no per-iter select /
// 64-bit muls), software-pipelined so next loads overlap current WMMAs.
__device__ __forceinline__ void gemm_seg(const bf16* __restrict__ src, int rs, int len,
                                         const bf16* __restrict__& wptr,
                                         int mg, int grp,
                                         v8f& acc0, v8f& acc1)
{
  if (len <= 0) return;
  const bf16* a0p = src + (size_t)mg * rs + grp * 8;
  const bf16* a1p = a0p + (size_t)16 * rs;
  const int goff = grp * 8;

  v16bf a0 = load_afrag(a0p);
  v16bf a1 = load_afrag(a1p);
  v16bf bb = load_bfrag(wptr + goff);
#pragma unroll 2
  for (int kb = 32; kb < len; kb += 32) {
    a0p += 32; a1p += 32; wptr += 32;
    v16bf na0 = load_afrag(a0p);
    v16bf na1 = load_afrag(a1p);
    v16bf nbb = load_bfrag(wptr + goff);
    acc0 = WMMA_BF16(a0, bb, acc0);
    acc1 = WMMA_BF16(a1, bb, acc1);
    a0 = na0; a1 = na1; bb = nbb;
  }
  acc0 = WMMA_BF16(a0, bb, acc0);
  acc1 = WMMA_BF16(a1, bb, acc1);
  wptr += 32;
}

// ------------------------------------------------------------------
// Generic M=32-per-block bf16 WMMA GEMM:  G[m][n] = sum_k A[m][k]*W[n][k] + bias[n]
// A = concat of up to 3 row-major sources (k0+k1+k2 == Ktot, each mult of 32).
// W row-major (N x Ktot) bf16. Each wave owns one 16-wide N strip, both 16-row
// halves of the 32-row M block (blockIdx.y).
// mode 0: f32 G (row stride Nt) | mode 1: mel/gate scatter into d_out | mode 2: bf16 relu
// ------------------------------------------------------------------
__global__ __launch_bounds__(128)
void gemm32_bf16(const bf16* __restrict__ s0, int k0,
                 const bf16* __restrict__ s1, int k1,
                 const bf16* __restrict__ s2, int k2,
                 const bf16* __restrict__ W,
                 const float* __restrict__ bias,
                 int Ktot, int Nt, int nstrips,
                 float* __restrict__ outF,
                 bf16* __restrict__ outB,
                 int mode, int step)
{
  const int wave  = threadIdx.x >> 5;
  const int lane  = threadIdx.x & 31;
  const int strip = blockIdx.x * 4 + wave;
  if (strip >= nstrips) return;
  const int n0   = strip * 16;
  const int mblk = blockIdx.y;
  const int nl   = lane & 15;
  const int grp  = lane >> 4;
  const int mg   = mblk * 32 + nl;   // A row index for tile0 (tile1 = +16)
  const int ncol = n0 + nl;

  v8f acc0 = {};
  v8f acc1 = {};

  const bf16* __restrict__ wptr = W + (size_t)ncol * Ktot;
  gemm_seg(s0, k0, k0, wptr, mg, grp, acc0, acc1);
  gemm_seg(s1, k1, k1, wptr, mg, grp, acc0, acc1);
  gemm_seg(s2, k2, k2, wptr, mg, grp, acc0, acc1);

  const float bv = bias ? bias[ncol] : 0.0f;
#pragma unroll
  for (int r = 0; r < 8; ++r) {
    const int ml = r + grp * 8;         // 0..15 within M tile
    float v0 = acc0[r] + bv;
    float v1 = acc1[r] + bv;
    if (mode == 0) {
      outF[(size_t)(mblk * 32 + ml) * Nt + ncol]      = v0;
      outF[(size_t)(mblk * 32 + ml + 16) * Nt + ncol] = v1;
    } else if (mode == 2) {
      outB[(size_t)(mblk * 32 + ml) * Nt + ncol]      = (bf16)fmaxf(v0, 0.0f);
      outB[(size_t)(mblk * 32 + ml + 16) * Nt + ncol] = (bf16)fmaxf(v1, 0.0f);
    } else { // mode 1: projection write-back into d_out, M rows are batches
      const int j = ncol;
      if (j < N_MEL * NF + NF) {
        const int b0 = ml, b1 = ml + 16;
        if (j < N_MEL * NF) {
          const int f = j / N_MEL, mm = j % N_MEL;
          outF[(size_t)b0 * (N_MEL * T_OUT) + mm * T_OUT + 2 * step + f] = v0;
          outF[(size_t)b1 * (N_MEL * T_OUT) + mm * T_OUT + 2 * step + f] = v1;
        } else {
          const int f = j - N_MEL * NF;
          outF[MEL_ELEMS + b0 * T_OUT + 2 * step + f] = v0;
          outF[MEL_ELEMS + b1 * T_OUT + 2 * step + f] = v1;
        }
      }
    }
  }
}

// ------------------------------------------------------------------
// LSTM pointwise (decoder cell): G (32x4096, gates i|f|g|o) -> c, h(bf16)
// ------------------------------------------------------------------
__global__ void lstm_pointwise(const float* __restrict__ G,
                               float* __restrict__ C,
                               bf16* __restrict__ Hbf)
{
  int idx = blockIdx.x * blockDim.x + threadIdx.x;
  if (idx >= BATCH * RNN) return;
  int b = idx >> 10, j = idx & 1023;
  const float* g = G + (size_t)b * 4 * RNN;
  float gi = g[j], gf = g[RNN + j], gg = g[2 * RNN + j], go = g[3 * RNN + j];
  float c  = C[idx];
  float c2 = sigf(gf) * c + sigf(gi) * tanhf(gg);
  float h  = sigf(go) * tanhf(c2);
  C[idx]   = c2;
  Hbf[idx] = (bf16)h;
}

// ------------------------------------------------------------------
// Fused attention step (block = batch):
//   att-LSTM pointwise -> h (LDS) ; query proj ; loc conv31 ; energies ;
//   softmax ; alignments ; cumulative weights ; context (bf16 out)
// ------------------------------------------------------------------
__global__ __launch_bounds__(256)
void attention_step(const float* __restrict__ Gatt,
                    float* __restrict__ attC,
                    bf16*  __restrict__ attHbf,
                    const float* __restrict__ w_query,
                    const float* __restrict__ lcw,     // (32,2,31)
                    const float* __restrict__ ldw,     // (128,32)
                    const float* __restrict__ vvec,    // (128)
                    const float* __restrict__ pm,      // (B,T_IN,128)
                    const float* __restrict__ memory,  // (B,T_IN,512)
                    float* __restrict__ aw,
                    float* __restrict__ awc,
                    bf16*  __restrict__ ctxBf,
                    float* __restrict__ dalign,
                    int step)
{
  __shared__ float s_aw[T_IN], s_awc[T_IN], s_e[T_IN];
  __shared__ float s_pq[ATT_DIM], s_h[RNN];
  __shared__ float s_lcw[LOC_F * 2 * LOC_K];   // 1984
  __shared__ float s_ldw[ATT_DIM * LOC_F];     // 4096
  __shared__ float s_v[ATT_DIM];
  __shared__ float red[256];

  const int b = blockIdx.x, tid = threadIdx.x;

  for (int i = tid; i < T_IN; i += 256) { s_aw[i] = aw[b * T_IN + i]; s_awc[i] = awc[b * T_IN + i]; }
  for (int i = tid; i < LOC_F * 2 * LOC_K; i += 256) s_lcw[i] = lcw[i];
  for (int i = tid; i < ATT_DIM * LOC_F; i += 256)   s_ldw[i] = ldw[i];
  if (tid < ATT_DIM) s_v[tid] = vvec[tid];

  // attention-LSTM pointwise for this batch
  for (int j = tid; j < RNN; j += 256) {
    const float* g = Gatt + (size_t)b * 4 * RNN;
    float gi = g[j], gf = g[RNN + j], gg = g[2 * RNN + j], go = g[3 * RNN + j];
    float c  = attC[b * RNN + j];
    float c2 = sigf(gf) * c + sigf(gi) * tanhf(gg);
    float h  = sigf(go) * tanhf(c2);
    attC[b * RNN + j]   = c2;
    attHbf[b * RNN + j] = (bf16)h;
    s_h[j] = h;
  }
  __syncthreads();

  // query projection pq = h @ w_query^T
  if (tid < ATT_DIM) {
    float acc = 0.0f;
    const float* wq = w_query + (size_t)tid * RNN;
    for (int k = 0; k < RNN; ++k) acc += wq[k] * s_h[k];
    s_pq[tid] = acc;
  }
  __syncthreads();

  // conv31 + energies (clamped tap bounds: no per-tap branch)
  for (int t = tid; t < T_IN; t += 256) {
    const int klo = (t < 15) ? (15 - t) : 0;
    const int khi = (t > T_IN - 1 - 15) ? (T_IN - 1 + 15 - t + 1) : LOC_K;
    float locf[LOC_F];
#pragma unroll 4
    for (int f = 0; f < LOC_F; ++f) {
      float a = 0.0f;
      const float* w0 = s_lcw + f * (2 * LOC_K);
      for (int k = klo; k < khi; ++k) {
        const int tt = t + k - 15;
        a += w0[k] * s_aw[tt] + w0[LOC_K + k] * s_awc[tt];
      }
      locf[f] = a;
    }
    float e = 0.0f;
    const float* pmr = pm + ((size_t)b * T_IN + t) * ATT_DIM;
    for (int d = 0; d < ATT_DIM; ++d) {
      float sd = s_pq[d] + pmr[d];
      const float* wr = s_ldw + d * LOC_F;
#pragma unroll
      for (int f = 0; f < LOC_F; ++f) sd += wr[f] * locf[f];
      e += s_v[d] * tanhf(sd);
    }
    s_e[t] = e;
  }
  __syncthreads();

  // softmax over T_IN
  float mx = -1e30f;
  for (int t = tid; t < T_IN; t += 256) mx = fmaxf(mx, s_e[t]);
  red[tid] = mx; __syncthreads();
  for (int o = 128; o > 0; o >>= 1) { if (tid < o) red[tid] = fmaxf(red[tid], red[tid + o]); __syncthreads(); }
  mx = red[0]; __syncthreads();
  float sm = 0.0f;
  for (int t = tid; t < T_IN; t += 256) { float ex = __expf(s_e[t] - mx); s_e[t] = ex; sm += ex; }
  red[tid] = sm; __syncthreads();
  for (int o = 128; o > 0; o >>= 1) { if (tid < o) red[tid] += red[tid + o]; __syncthreads(); }
  float inv = 1.0f / red[0];
  __syncthreads();

  for (int t = tid; t < T_IN; t += 256) {
    float a = s_e[t] * inv;
    s_aw[t] = a;
    float ac = s_awc[t] + a;
    aw[b * T_IN + t]  = a;
    awc[b * T_IN + t] = ac;
    dalign[((size_t)b * STEPS + step) * T_IN + t] = a;
  }
  __syncthreads();

  // context = aw . memory  -> bf16 for next GEMMs
  for (int e = tid; e < ENC; e += 256) {
    float acc = 0.0f;
    const float* mp = memory + (size_t)b * T_IN * ENC + e;
    for (int t = 0; t < T_IN; ++t) acc += s_aw[t] * mp[(size_t)t * ENC];
    ctxBf[b * ENC + e] = (bf16)acc;
  }
}

// ---------------- prep kernels ----------------
__global__ void k_concat2(bf16* __restrict__ dst, const float* __restrict__ a, int Ka,
                          const float* __restrict__ bsrc, int Kb, int N)
{
  const int K = Ka + Kb;
  const size_t total = (size_t)N * K;
  for (size_t i = (size_t)blockIdx.x * blockDim.x + threadIdx.x; i < total;
       i += (size_t)gridDim.x * blockDim.x) {
    int n = (int)(i / K), k = (int)(i % K);
    float v = (k < Ka) ? a[(size_t)n * Ka + k] : bsrc[(size_t)n * Kb + (k - Ka)];
    dst[i] = (bf16)v;
  }
}

__global__ void k_copy_bf(bf16* __restrict__ dst, const float* __restrict__ src, size_t n)
{
  for (size_t i = (size_t)blockIdx.x * blockDim.x + threadIdx.x; i < n;
       i += (size_t)gridDim.x * blockDim.x)
    dst[i] = (bf16)src[i];
}

__global__ void k_proj_pad(bf16* __restrict__ dst, const float* __restrict__ pw,
                           const float* __restrict__ gw)
{
  const int K = RNN + ENC;               // 1536
  const size_t total = (size_t)176 * K;
  for (size_t i = (size_t)blockIdx.x * blockDim.x + threadIdx.x; i < total;
       i += (size_t)gridDim.x * blockDim.x) {
    int n = (int)(i / K), k = (int)(i % K);
    float v = 0.0f;
    if (n < N_MEL * NF)           v = pw[(size_t)n * K + k];
    else if (n < N_MEL * NF + NF) v = gw[(size_t)(n - N_MEL * NF) * K + k];
    dst[i] = (bf16)v;
  }
}

__global__ void k_prep_state(float* attB, const float* abih, const float* abhh,
                             float* decB, const float* dbih, const float* dbhh,
                             float* projB, const float* pb, const float* gb,
                             float* attC, float* decC, bf16* attHbf, bf16* decHbf,
                             float* aw, float* awc)
{
  int i = blockIdx.x * blockDim.x + threadIdx.x;
  if (i < 4 * RNN) { attB[i] = abih[i] + abhh[i]; decB[i] = dbih[i] + dbhh[i]; }
  if (i < 176) projB[i] = (i < N_MEL * NF) ? pb[i] : ((i < N_MEL * NF + NF) ? gb[i - N_MEL * NF] : 0.0f);
  if (i < BATCH * RNN) { attC[i] = 0.0f; decC[i] = 0.0f; attHbf[i] = (bf16)0.0f; decHbf[i] = (bf16)0.0f; }
  if (i < BATCH * T_IN) { aw[i] = 0.0f; awc[i] = 0.0f; }
}

// di (STEPS*B, 160) bf16, row r = s*32+b, col c = f*80+m, shifted by one frame (go frame)
__global__ void k_build_di(bf16* __restrict__ di, const float* __restrict__ dec_in)
{
  const size_t total = (size_t)STEPS * BATCH * (N_MEL * NF);
  for (size_t i = (size_t)blockIdx.x * blockDim.x + threadIdx.x; i < total;
       i += (size_t)gridDim.x * blockDim.x) {
    int c = (int)(i % (N_MEL * NF));
    int r = (int)(i / (N_MEL * NF));
    int s = r / BATCH, b = r % BATCH;
    int f = c / N_MEL, m = c % N_MEL;
    float v = 0.0f;
    if (s > 0) v = dec_in[(size_t)b * N_MEL * T_OUT + (size_t)m * T_OUT + (s - 1) * 2 + f];
    di[i] = (bf16)v;
  }
}

// ------------------------------------------------------------------
extern "C" void kernel_launch(void* const* d_in, const int* in_sizes, int n_in,
                              void* d_out, int out_size, void* d_ws, size_t ws_size,
                              hipStream_t stream)
{
  const float* memory     = (const float*)d_in[0];
  const float* dec_inputs = (const float*)d_in[1];
  const float* prenet_w1  = (const float*)d_in[2];
  const float* prenet_w2  = (const float*)d_in[3];
  const float* att_w_ih   = (const float*)d_in[4];
  const float* att_w_hh   = (const float*)d_in[5];
  const float* att_b_ih   = (const float*)d_in[6];
  const float* att_b_hh   = (const float*)d_in[7];
  const float* w_query    = (const float*)d_in[8];
  const float* w_memory   = (const float*)d_in[9];
  const float* v          = (const float*)d_in[10];
  const float* loc_conv_w = (const float*)d_in[11];
  const float* loc_dense_w= (const float*)d_in[12];
  const float* dec_w_ih   = (const float*)d_in[13];
  const float* dec_w_hh   = (const float*)d_in[14];
  const float* dec_b_ih   = (const float*)d_in[15];
  const float* dec_b_hh   = (const float*)d_in[16];
  const float* proj_w     = (const float*)d_in[17];
  const float* proj_b     = (const float*)d_in[18];
  const float* gate_w     = (const float*)d_in[19];
  const float* gate_b     = (const float*)d_in[20];

  // ---- workspace carve (256B aligned) ----
  char* base = (char*)d_ws;
  size_t off = 0;
  auto carve = [&](size_t bytes) -> void* {
    void* p = base + off;
    off = (off + bytes + 255) & ~(size_t)255;
    return p;
  };
  bf16* attW   = (bf16*)carve((size_t)4 * RNN * (PRENET + RNN) * 2);      // 4096x1280
  bf16* decW   = (bf16*)carve((size_t)4 * RNN * (PRENET + ENC + RNN) * 2);// 4096x1792
  bf16* projW  = (bf16*)carve((size_t)176 * (RNN + ENC) * 2);             // 176x1536
  bf16* w1bf   = (bf16*)carve((size_t)PRENET * (N_MEL * NF) * 2);
  bf16* w2bf   = (bf16*)carve((size_t)PRENET * PRENET * 2);
  bf16* wmemBf = (bf16*)carve((size_t)ATT_DIM * ENC * 2);
  bf16* membf  = (bf16*)carve((size_t)BATCH * T_IN * ENC * 2);
  float* attB  = (float*)carve((size_t)4 * RNN * 4);
  float* decB  = (float*)carve((size_t)4 * RNN * 4);
  float* projB = (float*)carve((size_t)176 * 4);
  bf16* di     = (bf16*)carve((size_t)STEPS * BATCH * (N_MEL * NF) * 2);
  bf16* pre1   = (bf16*)carve((size_t)STEPS * BATCH * PRENET * 2);
  bf16* xs     = (bf16*)carve((size_t)STEPS * BATCH * PRENET * 2);
  float* pm    = (float*)carve((size_t)BATCH * T_IN * ATT_DIM * 4);
  float* Gatt  = (float*)carve((size_t)BATCH * 4 * RNN * 4);
  float* Gdec  = (float*)carve((size_t)BATCH * 4 * RNN * 4);
  bf16* attHbf = (bf16*)carve((size_t)BATCH * RNN * 2);
  float* attC  = (float*)carve((size_t)BATCH * RNN * 4);
  bf16* decHbf = (bf16*)carve((size_t)BATCH * RNN * 2);
  float* decC  = (float*)carve((size_t)BATCH * RNN * 4);
  float* awBuf = (float*)carve((size_t)BATCH * T_IN * 4);
  float* awcBuf= (float*)carve((size_t)BATCH * T_IN * 4);
  bf16* ctxBf  = (bf16*)carve((size_t)BATCH * ENC * 2);
  (void)ws_size; (void)in_sizes; (void)n_in; (void)out_size;

  float* dout   = (float*)d_out;
  float* dalign = dout + MEL_ELEMS + GATE_ELEMS;

  // ---- one-time prep ----
  k_concat2<<<256, 256, 0, stream>>>(attW, att_w_ih, PRENET, att_w_hh, RNN, 4 * RNN);
  k_concat2<<<256, 256, 0, stream>>>(decW, dec_w_ih, PRENET + ENC, dec_w_hh, RNN, 4 * RNN);
  k_proj_pad<<<256, 256, 0, stream>>>(projW, proj_w, gate_w);
  k_copy_bf<<<64, 256, 0, stream>>>(w1bf, prenet_w1, (size_t)PRENET * N_MEL * NF);
  k_copy_bf<<<64, 256, 0, stream>>>(w2bf, prenet_w2, (size_t)PRENET * PRENET);
  k_copy_bf<<<64, 256, 0, stream>>>(wmemBf, w_memory, (size_t)ATT_DIM * ENC);
  k_copy_bf<<<2048, 256, 0, stream>>>(membf, memory, (size_t)BATCH * T_IN * ENC);
  k_prep_state<<<128, 256, 0, stream>>>(attB, att_b_ih, att_b_hh, decB, dec_b_ih, dec_b_hh,
                                        projB, proj_b, gate_b, attC, decC, attHbf, decHbf,
                                        awBuf, awcBuf);
  k_build_di<<<2048, 256, 0, stream>>>(di, dec_inputs);

  // processed_memory = memory @ w_memory^T  (12800x128, K=512)
  gemm32_bf16<<<dim3(2, (BATCH * T_IN) / 32), 128, 0, stream>>>(
      membf, ENC, membf, 0, membf, 0, wmemBf, nullptr,
      ENC, ATT_DIM, ATT_DIM / 16, pm, nullptr, 0, 0);

  // prenet: relu(relu(di @ w1^T) @ w2^T)  (9600 rows)
  gemm32_bf16<<<dim3(4, STEPS), 128, 0, stream>>>(
      di, N_MEL * NF, di, 0, di, 0, w1bf, nullptr,
      N_MEL * NF, PRENET, PRENET / 16, nullptr, pre1, 2, 0);
  gemm32_bf16<<<dim3(4, STEPS), 128, 0, stream>>>(
      pre1, PRENET, pre1, 0, pre1, 0, w2bf, nullptr,
      PRENET, PRENET, PRENET / 16, nullptr, xs, 2, 0);

  // ---- 300 sequential decoder steps ----
  for (int s = 0; s < STEPS; ++s) {
    const bf16* x_t = xs + (size_t)s * BATCH * PRENET;

    // attention LSTM gates: [x | h] (K=1280) @ attW^T + bias
    gemm32_bf16<<<dim3(64, 1), 128, 0, stream>>>(
        x_t, PRENET, attHbf, RNN, attHbf, 0, attW, attB,
        PRENET + RNN, 4 * RNN, 4 * RNN / 16, Gatt, nullptr, 0, 0);

    // fused attention (pointwise LSTM + location attention + context)
    attention_step<<<BATCH, 256, 0, stream>>>(
        Gatt, attC, attHbf, w_query, loc_conv_w, loc_dense_w, v,
        pm, memory, awBuf, awcBuf, ctxBf, dalign, s);

    // decoder LSTM gates: [x | ctx | h] (K=1792) @ decW^T + bias
    gemm32_bf16<<<dim3(64, 1), 128, 0, stream>>>(
        x_t, PRENET, ctxBf, ENC, decHbf, RNN, decW, decB,
        PRENET + ENC + RNN, 4 * RNN, 4 * RNN / 16, Gdec, nullptr, 0, 0);

    lstm_pointwise<<<128, 256, 0, stream>>>(Gdec, decC, decHbf);

    // projection + gate: [dec_h | ctx] (K=1536) -> mel/gate scatter into d_out
    gemm32_bf16<<<dim3(3, 1), 128, 0, stream>>>(
        decHbf, RNN, ctxBf, ENC, ctxBf, 0, projW, projB,
        RNN + ENC, 176, 11, dout, nullptr, 1, s);
  }
}
